// PNN_84043920048524
// MI455X (gfx1250) — compile-verified, matrix operand order checked by reading, not run on hardware
//
#include <hip/hip_runtime.h>
#include <hip/hip_bf16.h>
#include <math.h>

// ---------------------------------------------------------------------------
// PNN (IPNN) inference for MI455X / gfx1250, wave32, bf16 WMMA w/ f32 accum.
// B=4096, N=39 fields, M=32 emb dim, D=32 product dim, MLP 64->400->400->400->1
// ---------------------------------------------------------------------------

#define BATCH   4096
#define NFIELD  39
#define EMB     32
#define DPROD   32
#define H_DIM   400
#define BN_EPS  1e-3f

typedef __attribute__((ext_vector_type(16))) __bf16 v16bf;
typedef __attribute__((ext_vector_type(16))) float  v16f;
typedef __attribute__((ext_vector_type(8)))  float  v8f;
typedef int v4i_ __attribute__((vector_size(4 * sizeof(int))));

#define AS1 __attribute__((address_space(1)))
#define AS3 __attribute__((address_space(3)))

// Async global->LDS path (gfx1250): gate on builtin availability so the host
// pass / older toolchains fall back to synchronous copies.
#if defined(__AMDGCN__) && \
    __has_builtin(__builtin_amdgcn_global_load_async_to_lds_b128) && \
    __has_builtin(__builtin_amdgcn_s_wait_asynccnt)
#define PNN_ASYNC 1
#else
#define PNN_ASYNC 0
#endif

// ---------------------------------------------------------------------------
// Kernel 1: embedding gather + product layer.
//   x0[b, 0:32]  = l_z[b, :]   (e · lin_w)
//   x0[b,32:64]  = l_p[b, :]   (= sum_n quad_w[d,n]^2 * ||e[b,n,:]||^2)
// One wave per 16-row batch tile; per field n (K-chunk of 32) two
// v_wmma_f32_16x16x32_bf16 issues cover the d=0..31 output.
//
// 16-bit A-operand layout (ISA 7.12.2): lane<16 holds row=lane, K={0..7,16..23};
// lane>=16 holds row=lane-16, K={8..15,24..31}.  B loaded as B^T, mirrored.
// ---------------------------------------------------------------------------
__global__ __launch_bounds__(128)
void pnn_product_kernel(const int*   __restrict__ fidx,
                        const float* __restrict__ emb,
                        const float* __restrict__ lin_w,
                        const float* __restrict__ quad_w,
                        __bf16*      __restrict__ x0) {
  __shared__ float sh_qw2[DPROD * NFIELD];      // quad_w^2, 5 KB
  const int tid = threadIdx.x;
  for (int i = tid; i < DPROD * NFIELD; i += 128) {
    float q = quad_w[i];
    sh_qw2[i] = q * q;
  }
  __syncthreads();

  const int wave    = tid >> 5;
  const int lane    = tid & 31;
  const int half    = lane >> 4;                // 0: low K half, 1: high K half
  const int r       = lane & 15;
  const int rowBase = blockIdx.x * 64 + wave * 16;
  const int row     = rowBase + r;
  const int dbase   = half * 16;

  v8f c0 = {};                                   // l_z cols 0..15
  v8f c1 = {};                                   // l_z cols 16..31
  float acc_lp[16];
#pragma unroll
  for (int j = 0; j < 16; ++j) acc_lp[j] = 0.f;

  for (int n = 0; n < NFIELD; ++n) {
    const int idx = fidx[row * NFIELD + n];
    if (n + 1 < NFIELD) {                        // global_prefetch_b8 next row
      const int idx2 = fidx[row * NFIELD + n + 1];
      __builtin_prefetch(emb + (size_t)idx2 * EMB, 0, 1);
    }

    // Gather this lane's 16 embedding values (K pattern per A layout).
    const float4* e4 = (const float4*)(emb + (size_t)idx * EMB);
    float4 p0 = e4[half * 2 + 0];                // K = half*8 + 0..3
    float4 p1 = e4[half * 2 + 1];                // K = half*8 + 4..7
    float4 p2 = e4[4 + half * 2 + 0];            // K = 16 + half*8 + 0..3
    float4 p3 = e4[4 + half * 2 + 1];            // K = 16 + half*8 + 4..7
    v16f vv = {p0.x, p0.y, p0.z, p0.w, p1.x, p1.y, p1.z, p1.w,
               p2.x, p2.y, p2.z, p2.w, p3.x, p3.y, p3.z, p3.w};

    // ||e[b,n,:]||^2 : lane pair (L, L+16) holds the complementary K halves.
    float ss = 0.f;
#pragma unroll
    for (int i = 0; i < 16; ++i) ss += vv[i] * vv[i];
    ss += __shfl_xor(ss, 16);

#pragma unroll
    for (int j = 0; j < 16; ++j)
      acc_lp[j] += sh_qw2[(dbase + j) * NFIELD + n] * ss;

    v16bf a = __builtin_convertvector(vv, v16bf);   // v_cvt_pk_bf16_f32

    // B fragments: lin_w[d, n, m], lane column d = r (tile0) / r+16 (tile1).
    const float4* w4a = (const float4*)(lin_w + (size_t)r        * (NFIELD * EMB) + n * EMB);
    const float4* w4b = (const float4*)(lin_w + (size_t)(r + 16) * (NFIELD * EMB) + n * EMB);
    float4 q0 = w4a[half * 2 + 0], q1 = w4a[half * 2 + 1];
    float4 q2 = w4a[4 + half * 2 + 0], q3 = w4a[4 + half * 2 + 1];
    v16f wv = {q0.x, q0.y, q0.z, q0.w, q1.x, q1.y, q1.z, q1.w,
               q2.x, q2.y, q2.z, q2.w, q3.x, q3.y, q3.z, q3.w};
    v16bf b0 = __builtin_convertvector(wv, v16bf);

    q0 = w4b[half * 2 + 0]; q1 = w4b[half * 2 + 1];
    q2 = w4b[4 + half * 2 + 0]; q3 = w4b[4 + half * 2 + 1];
    v16f wu = {q0.x, q0.y, q0.z, q0.w, q1.x, q1.y, q1.z, q1.w,
               q2.x, q2.y, q2.z, q2.w, q3.x, q3.y, q3.z, q3.w};
    v16bf b1 = __builtin_convertvector(wu, v16bf);

    c0 = __builtin_amdgcn_wmma_f32_16x16x32_bf16(false, a, false, b0, (short)0, c0, false, false);
    c1 = __builtin_amdgcn_wmma_f32_16x16x32_bf16(false, a, false, b1, (short)0, c1, false, false);
  }

  // C layout: lane L, VGPR j -> row m = j + 8*(L>=16), col = L&15.
#pragma unroll
  for (int j = 0; j < 8; ++j) {
    const int m = j + 8 * half;
    x0[(size_t)(rowBase + m) * 64 + r]      = (__bf16)c0[j];
    x0[(size_t)(rowBase + m) * 64 + 16 + r] = (__bf16)c1[j];
  }
#pragma unroll
  for (int j = 0; j < 16; ++j)
    x0[(size_t)row * 64 + 32 + dbase + j] = (__bf16)acc_lp[j];
}

// ---------------------------------------------------------------------------
// Kernels 2-4: dense layer  out = relu( BN( in @ W + b ) ), BN folded into
// per-column scale/shift.  Block = 256 threads (8 waves), block output tile =
// 16 rows x 128 cols.  The 32(k) x 128(col) fp32 W panel for each K-chunk is
// double-buffered into LDS with GLOBAL_LOAD_ASYNC_TO_LDS_B128 (ASYNCcnt),
// overlapping the WMMA of the previous chunk.
// ---------------------------------------------------------------------------
__device__ __forceinline__
void pnn_stage_w(const float* __restrict__ W, int Kin, int Kout,
                 int kb, int cBase, float* dst, float* dummy, int tid) {
#pragma unroll
  for (int i = 0; i < 4; ++i) {
    const int t    = i * 256 + tid;        // 0..1023 transfer slots
    const int krow = t >> 5;               // 0..31   k row within chunk
    const int lc   = t & 31;               // 16-byte column group
    const bool ok  = ((kb + krow) < Kin) && ((cBase + lc * 4) < Kout);
    // Always issue exactly 4 transfers per thread (uniform ASYNCcnt), clamping
    // out-of-range slots to a safe source and a dummy LDS sink.
    const float* g = ok ? (W + (size_t)(kb + krow) * Kout + cBase + lc * 4) : W;
    float*       l = ok ? (dst + krow * 128 + lc * 4) : dummy;
#if PNN_ASYNC
    __builtin_amdgcn_global_load_async_to_lds_b128((v4i_*)g, (AS3 v4i_*)l, 0, 0);
#else
    const float4 v = *(const float4*)g;
    *(float4*)l = v;
#endif
  }
}

__global__ __launch_bounds__(256)
void pnn_mlp_layer_kernel(const __bf16* __restrict__ in, int Kin,
                          const float* __restrict__ W,
                          const float* __restrict__ bias,
                          const float* __restrict__ gamma,
                          const float* __restrict__ beta,
                          const float* __restrict__ rmean,
                          const float* __restrict__ rvar,
                          __bf16* __restrict__ out, int Kout) {
  __shared__ __align__(16) float shW[2][32 * 128];   // double-buffered W panel
  __shared__ __align__(16) float shDummy[4];

  const int tid     = threadIdx.x;
  const int lane    = tid & 31;
  const int wave    = tid >> 5;
  const int colTile = blockIdx.y * 8 + wave;
  const bool active = (colTile * 16) < Kout;         // wave-uniform
  const int half    = lane >> 4;
  const int r       = lane & 15;
  const int rowBase = blockIdx.x * 16;
  const int cBase   = blockIdx.y * 128;
  const int colLoc  = wave * 16 + r;                 // column within LDS panel
  const int col     = cBase + colLoc;

  const int nChunks = (Kin + 31) / 32;

  pnn_stage_w(W, Kin, Kout, 0, cBase, shW[0], shDummy, tid);

  v8f acc = {};
  for (int kc = 0; kc < nChunks; ++kc) {
    if (kc + 1 < nChunks) {
      pnn_stage_w(W, Kin, Kout, (kc + 1) * 32, cBase, shW[(kc + 1) & 1], shDummy, tid);
#if PNN_ASYNC
      __builtin_amdgcn_s_wait_asynccnt(4);   // retire chunk kc, leave kc+1 inflight
#endif
    } else {
#if PNN_ASYNC
      __builtin_amdgcn_s_wait_asynccnt(0);
#endif
    }
    __syncthreads();                          // chunk kc visible to all waves

    if (active) {
      const int kb = kc * 32;
      const float* wp = shW[kc & 1];
      v16bf a;
      v16f  bf_ = {};
#pragma unroll
      for (int i = 0; i < 16; ++i) {
        const int koff = (i < 8) ? (half * 8 + i) : (16 + half * 8 + (i - 8));
        const int k = kb + koff;
        if (k < Kin) {
          a[i]   = in[(size_t)(rowBase + r) * Kin + k];
          bf_[i] = wp[koff * 128 + colLoc];
        } else {
          a[i] = (__bf16)0.f;
        }
      }
      const v16bf b = __builtin_convertvector(bf_, v16bf);
      acc = __builtin_amdgcn_wmma_f32_16x16x32_bf16(false, a, false, b, (short)0, acc, false, false);
    }
    __syncthreads();                          // reads done before buffer reuse
  }

  if (active) {
    const float s  = gamma[col] * rsqrtf(rvar[col] + BN_EPS);
    const float sh = (bias[col] - rmean[col]) * s + beta[col];
#pragma unroll
    for (int j = 0; j < 8; ++j) {
      const int m = j + 8 * half;
      float v = acc[j] * s + sh;
      v = v > 0.f ? v : 0.f;                   // ReLU
      out[(size_t)(rowBase + m) * Kout + col] = (__bf16)v;
    }
  }
}

// ---------------------------------------------------------------------------
// Kernel 5: sigmoid( h3 @ Wout + bout )  -> fp32 (B,1)
// ---------------------------------------------------------------------------
__global__ __launch_bounds__(256)
void pnn_out_kernel(const __bf16* __restrict__ h,
                    const float* __restrict__ Wout,
                    const float* __restrict__ bout,
                    float* __restrict__ out) {
  const int row = blockIdx.x * blockDim.x + threadIdx.x;
  if (row >= BATCH) return;
  const __bf16* hr = h + (size_t)row * H_DIM;
  float s = 0.f;
#pragma unroll 8
  for (int k = 0; k < H_DIM; ++k) s += (float)hr[k] * Wout[k];
  s += bout[0];
  out[row] = 1.f / (1.f + expf(-s));
}

// ---------------------------------------------------------------------------
extern "C" void kernel_launch(void* const* d_in, const int* in_sizes, int n_in,
                              void* d_out, int out_size, void* d_ws, size_t ws_size,
                              hipStream_t stream) {
  const int*   fidx   = (const int*)  d_in[0];
  // d_in[1] = feature_value: unused by the reference model
  const float* emb    = (const float*)d_in[2];
  const float* lin_w  = (const float*)d_in[3];
  const float* quad_w = (const float*)d_in[4];

  const float* W0  = (const float*)d_in[5];
  const float* b0  = (const float*)d_in[6];
  const float* g0  = (const float*)d_in[7];
  const float* be0 = (const float*)d_in[8];
  const float* rm0 = (const float*)d_in[9];
  const float* rv0 = (const float*)d_in[10];
  const float* W1  = (const float*)d_in[11];
  const float* b1  = (const float*)d_in[12];
  const float* g1  = (const float*)d_in[13];
  const float* be1 = (const float*)d_in[14];
  const float* rm1 = (const float*)d_in[15];
  const float* rv1 = (const float*)d_in[16];
  const float* W2  = (const float*)d_in[17];
  const float* b2  = (const float*)d_in[18];
  const float* g2  = (const float*)d_in[19];
  const float* be2 = (const float*)d_in[20];
  const float* rm2 = (const float*)d_in[21];
  const float* rv2 = (const float*)d_in[22];
  const float* Wout = (const float*)d_in[23];
  const float* bout = (const float*)d_in[24];

  char* ws = (char*)d_ws;
  __bf16* x0 = (__bf16*)ws;                               // 4096*64   bf16
  __bf16* h1 = (__bf16*)(ws + (size_t)BATCH * 64 * 2);    // 4096*400  bf16
  __bf16* h2 = h1 + (size_t)BATCH * H_DIM;
  __bf16* h3 = h2 + (size_t)BATCH * H_DIM;

  pnn_product_kernel<<<BATCH / 64, 128, 0, stream>>>(fidx, emb, lin_w, quad_w, x0);

  dim3 grid(BATCH / 16, 4);   // 25 col tiles of 16 over 8 waves -> grid.y = 4
  pnn_mlp_layer_kernel<<<grid, 256, 0, stream>>>(x0,  64,   W0, b0, g0, be0, rm0, rv0, h1, H_DIM);
  pnn_mlp_layer_kernel<<<grid, 256, 0, stream>>>(h1, H_DIM, W1, b1, g1, be1, rm1, rv1, h2, H_DIM);
  pnn_mlp_layer_kernel<<<grid, 256, 0, stream>>>(h2, H_DIM, W2, b2, g2, be2, rm2, rv2, h3, H_DIM);

  pnn_out_kernel<<<BATCH / 256, 256, 0, stream>>>(h3, Wout, bout, (float*)d_out);
}
